// ManualAttentionGRU_85031762526355
// MI455X (gfx1250) — matrix operand order repaired and checked by reading, compiled, not run on hardware
//
#include <hip/hip_runtime.h>
#include <hip/hip_bf16.h>

// ---------------------------------------------------------------------------
// ManualAttentionGRU on MI455X (gfx1250, wave32, WMMA bf16 16x16x32 + TDM)
// ---------------------------------------------------------------------------

typedef __attribute__((ext_vector_type(16))) __bf16 v16bf;
typedef __attribute__((ext_vector_type(8)))  __bf16 v8bf;
typedef __attribute__((ext_vector_type(8)))  float  v8f;
typedef __attribute__((ext_vector_type(4)))  unsigned int u32x4;
typedef __attribute__((ext_vector_type(8)))  int i32x8;
typedef __attribute__((ext_vector_type(4)))  int i32x4;

#define V_  32000
#define E_  128
#define H_  128
#define B_  32
#define T_  128
#define G3  384     // 3*H

union V16U { v16bf v; v8bf h[2]; };

__device__ __forceinline__ v8f wmma_bf16(v16bf a, v16bf b, v8f c) {
  return __builtin_amdgcn_wmma_f32_16x16x32_bf16(
      false, a, false, b, (short)0, c, false, false);
}

// A fragment (16x32 bf16, wave32): lane m=lane&15; contiguous K:
//   k = kb + (lane>=16)*16 + e
__device__ __forceinline__ int a_koff(int half, int e) {
  return (half << 4) + e;
}

// A fragment from a bf16 row (16B-aligned runs) -> two b128 loads
__device__ __forceinline__ v16bf load_afrag_bf(const __bf16* row, int kb, int half) {
  V16U u;
  u.h[0] = *(const v8bf*)(row + kb + half * 16);
  u.h[1] = *(const v8bf*)(row + kb + half * 16 + 8);
  return u.v;
}

// B fragment (32x16 bf16): lane n=lane&15; k = kb + (lane>=16)*16 + e
// from a K-contiguous transposed row -> two b128 loads
__device__ __forceinline__ v16bf load_bfrag(const __bf16* p) {
  V16U u;
  u.h[0] = *(const v8bf*)(p);
  u.h[1] = *(const v8bf*)(p + 8);
  return u.v;
}

// A fragment from f32 row with on-the-fly convert (small matrices only)
__device__ __forceinline__ v16bf cvt_afrag_f32(const float* row, int kb, int half) {
  v16bf a;
#pragma unroll
  for (int e = 0; e < 16; ++e) a[e] = (__bf16)row[kb + a_koff(half, e)];
  return a;
}

// ---------------------------------------------------------------------------
// TDM: 2D bf16 tile load Global -> LDS (D# per CDNA5 ISA)
// ---------------------------------------------------------------------------
__device__ __forceinline__ void tdm_load_2d_bf16(unsigned lds_off, const void* gptr,
                                                 unsigned tensor_d0, unsigned tensor_d1,
                                                 unsigned tile_d0, unsigned tile_d1,
                                                 unsigned stride_d0_elems) {
#if __has_builtin(__builtin_amdgcn_tensor_load_to_lds)
  unsigned long long ga = (unsigned long long)gptr;
  u32x4 g0;
  g0[0] = 1u;                                        // count=1 valid descriptor
  g0[1] = lds_off;                                   // lds_addr (bytes)
  g0[2] = (unsigned)(ga & 0xFFFFFFFFull);            // global_addr[31:0]
  g0[3] = (unsigned)((ga >> 32) & 0x01FFFFFFull)     // global_addr[56:32]
          | (2u << 30);                              // type=2 ("image")
  i32x8 g1;
  g1[0] = (int)(1u << 16);                           // data_size=1 -> 2 bytes
  g1[1] = (int)((tensor_d0 & 0xFFFFu) << 16);        // tensor_dim0[15:0] @bits63:48
  g1[2] = (int)(((tensor_d0 >> 16) & 0xFFFFu) | ((tensor_d1 & 0xFFFFu) << 16));
  g1[3] = (int)(((tensor_d1 >> 16) & 0xFFFFu) | ((tile_d0 & 0xFFFFu) << 16));
  g1[4] = (int)(tile_d1 & 0xFFFFu);                  // tile_dim1, tile_dim2=0
  g1[5] = (int)stride_d0_elems;                      // tensor_dim0_stride[31:0]
  g1[6] = 0;                                         // stride hi / dim1_stride
  g1[7] = 0;
  i32x4 gz = {0, 0, 0, 0};
#if defined(__clang_major__) && (__clang_major__ >= 23)
  i32x8 gz8 = {0, 0, 0, 0, 0, 0, 0, 0};
  __builtin_amdgcn_tensor_load_to_lds(g0, g1, gz, gz, gz8, 0);
#else
  __builtin_amdgcn_tensor_load_to_lds(g0, g1, gz, gz, 0);
#endif
#endif
}

__device__ __forceinline__ unsigned lds_offset_of(const void* p) {
  // generic pointers to LDS carry the LDS byte offset in addr[31:0]
  return (unsigned)(unsigned long long)p;
}

// ---------------------------------------------------------------------------
// K0a: transpose+convert small weights to bf16:  WT[n*K + k] = W[k*N + n]
// ---------------------------------------------------------------------------
__global__ void convert_small_kernel(const float* __restrict__ Wih,
                                     const float* __restrict__ Whh,
                                     const float* __restrict__ aw,
                                     __bf16* __restrict__ WihT,
                                     __bf16* __restrict__ WhhT,
                                     __bf16* __restrict__ awT) {
  int i = blockIdx.x * blockDim.x + threadIdx.x;
  if (i < G3 * E_) {
    int n = i >> 7, k = i & 127;
    WihT[i] = (__bf16)Wih[k * G3 + n];
    WhhT[i] = (__bf16)Whh[k * G3 + n];
  }
  if (i < H_ * H_) {
    int n = i >> 7, k = i & 127;
    awT[i] = (__bf16)aw[k * H_ + n];
  }
}

// K0b: fc_wT[n*256 + k] = bf16(fc_w[k*V + n])   (one-time 33MB->16MB, L2-resident)
__global__ void convert_fcw_kernel(const float* __restrict__ fcw,
                                   __bf16* __restrict__ fcwT) {
  long long i = (long long)blockIdx.x * blockDim.x + threadIdx.x;
  int n = (int)(i >> 8);
  int k = (int)(i & 255);
  __builtin_prefetch(&fcw[(long long)k * V_ + n + 256], 0, 1);
  fcwT[i] = (__bf16)fcw[(long long)k * V_ + n];
}

// ---------------------------------------------------------------------------
// K1: xg = embedding[x] @ W_ih + b   (M=4096, N=384, K=128), 1 wave / 16x16 tile
// ---------------------------------------------------------------------------
__global__ void embed_gates_kernel(const int* __restrict__ x,
                                   const float* __restrict__ emb,
                                   const __bf16* __restrict__ WihT,
                                   const float* __restrict__ bias,
                                   float* __restrict__ xg) {
  const int lane = threadIdx.x & 31;
  const int wave = threadIdx.x >> 5;
  const int half = lane >> 4;
  const int tile = blockIdx.x * 4 + wave;
  const int mt = tile / (G3 / 16);
  const int nt = tile % (G3 / 16);
  const int m0 = mt * 16;
  const int ncol = nt * 16 + (lane & 15);
  const float* arow = emb + (long long)x[m0 + (lane & 15)] * E_;
  const __bf16* brow = WihT + ncol * E_;

  v8f c = {};
#pragma unroll
  for (int kb = 0; kb < E_; kb += 32) {
    v16bf a = cvt_afrag_f32(arow, kb, half);
    v16bf b = load_bfrag(brow + kb + half * 16);
    c = wmma_bf16(a, b, c);
  }
  const float bn = bias[ncol];
#pragma unroll
  for (int r = 0; r < 8; ++r) {
    int mi = r + 8 * half;
    xg[(long long)(m0 + mi) * G3 + ncol] = c[r] + bn;
  }
}

// ---------------------------------------------------------------------------
// K2: GRU scan, sequential in T, 16 batch rows / block; h in LDS, W_hhT in L2.
// Writes both hs (f32, for softmax-context) and hsb (bf16, for attn GEMM / fc A)
// ---------------------------------------------------------------------------
__global__ void gru_scan_kernel(const float* __restrict__ xg,
                                const __bf16* __restrict__ WhhT,
                                float* __restrict__ hs,
                                __bf16* __restrict__ hsb) {
  __shared__ float h_lds[16 * H_];    // 8 KB
  __shared__ float g_lds[16 * G3];    // 24 KB
  const int lane = threadIdx.x & 31;
  const int wave = threadIdx.x >> 5;
  const int half = lane >> 4;
  const int bb0  = blockIdx.x * 16;

  for (int i = threadIdx.x; i < 16 * H_; i += blockDim.x) h_lds[i] = 0.f;
  __syncthreads();

  for (int t = 0; t < T_; ++t) {
    for (int nt = wave; nt < G3 / 16; nt += 8) {
      const int ncol = nt * 16 + (lane & 15);
      const __bf16* brow = WhhT + ncol * H_;
      const float* arow = h_lds + (lane & 15) * H_;
      v8f c = {};
#pragma unroll
      for (int kb = 0; kb < H_; kb += 32) {
        v16bf a = cvt_afrag_f32(arow, kb, half);
        v16bf b = load_bfrag(brow + kb + half * 16);
        c = wmma_bf16(a, b, c);
      }
#pragma unroll
      for (int r = 0; r < 8; ++r) {
        int mi = r + 8 * half;
        g_lds[mi * G3 + ncol] =
            c[r] + xg[((long long)(bb0 + mi) * T_ + t) * G3 + ncol];
      }
    }
    __syncthreads();
    for (int i = threadIdx.x; i < 16 * H_; i += blockDim.x) {
      int mi = i >> 7;
      int n  = i & (H_ - 1);
      float iz  = g_lds[mi * G3 + n];
      float ir  = g_lds[mi * G3 + H_ + n];
      float in_ = g_lds[mi * G3 + 2 * H_ + n];
      float z  = 1.f / (1.f + __expf(-iz));
      float r  = 1.f / (1.f + __expf(-ir));
      float nn = tanhf(in_ * r);
      float hn = (1.f - z) * nn + z * h_lds[i];
      h_lds[i] = hn;
      long long row = (long long)(bb0 + mi) * T_ + t;
      hs [row * H_ + n] = hn;
      hsb[row * H_ + n] = (__bf16)hn;
    }
    __syncthreads();
  }
}

// ---------------------------------------------------------------------------
// K3: scores = tanh(hs @ attn_w + attn_b) @ v_w + v_b (fused, LDS f32 atomics)
// ---------------------------------------------------------------------------
__global__ void attn_score_kernel(const __bf16* __restrict__ hsb,
                                  const __bf16* __restrict__ awT,
                                  const float* __restrict__ attn_b,
                                  const float* __restrict__ v_w,
                                  const float* __restrict__ v_b,
                                  float* __restrict__ scores) {
  __shared__ float acc[16];
  const int lane = threadIdx.x & 31;
  const int wave = threadIdx.x >> 5;
  const int half = lane >> 4;
  const int m0   = blockIdx.x * 16;
  const int ncol = wave * 16 + (lane & 15);

  if (threadIdx.x < 16) acc[threadIdx.x] = 0.f;
  __syncthreads();

  const __bf16* arow = hsb + (long long)(m0 + (lane & 15)) * H_;
  const __bf16* brow = awT + ncol * H_;
  v8f c = {};
#pragma unroll
  for (int kb = 0; kb < H_; kb += 32) {
    v16bf a = load_afrag_bf(arow, kb, half);
    v16bf b = load_bfrag(brow + kb + half * 16);
    c = wmma_bf16(a, b, c);
  }
  const float ab = attn_b[ncol];
  const float vw = v_w[ncol];
#pragma unroll
  for (int r = 0; r < 8; ++r) {
    int mi = r + 8 * half;
    atomicAdd(&acc[mi], tanhf(c[r] + ab) * vw);   // ds_add_f32
  }
  __syncthreads();
  if (threadIdx.x < 16) scores[m0 + threadIdx.x] = acc[threadIdx.x] + v_b[0];
}

// ---------------------------------------------------------------------------
// K4: causal softmax + context; emits fc A-matrix rows Abf = bf16[hs | ctx]
// one wave per (b,t); lane owns 4 H-components
// ---------------------------------------------------------------------------
__global__ void attn_context_kernel(const float* __restrict__ scores,
                                    const float* __restrict__ hs,
                                    const __bf16* __restrict__ hsb,
                                    __bf16* __restrict__ Abf) {
  const int lane = threadIdx.x & 31;
  const int g = blockIdx.x * 8 + (threadIdx.x >> 5);
  const int b = g >> 7;
  const int t = g & (T_ - 1);
  const float* sc = scores + b * T_;
  float cx = 0.f, cy = 0.f, cz = 0.f, cw = 0.f;
  if (t > 0) {
    float mx = -1e30f;
    for (int s = lane; s < t; s += 32) mx = fmaxf(mx, sc[s]);
#pragma unroll
    for (int off = 16; off > 0; off >>= 1) mx = fmaxf(mx, __shfl_xor(mx, off, 32));
    float sum = 0.f;
    for (int s = lane; s < t; s += 32) sum += __expf(sc[s] - mx);
#pragma unroll
    for (int off = 16; off > 0; off >>= 1) sum += __shfl_xor(sum, off, 32);
    const float inv = 1.f / sum;
    for (int s = 0; s < t; ++s) {
      float w = __expf(sc[s] - mx) * inv;
      const float* hr = hs + ((long long)b * T_ + s) * H_;
      cx += w * hr[lane];
      cy += w * hr[lane + 32];
      cz += w * hr[lane + 64];
      cw += w * hr[lane + 96];
    }
  }
  __bf16* ar = Abf + (long long)g * 256;
  const __bf16* hb = hsb + (long long)g * H_;
  ar[lane]      = hb[lane];
  ar[lane + 32] = hb[lane + 32];
  ar[lane + 64] = hb[lane + 64];
  ar[lane + 96] = hb[lane + 96];
  ar[128 + lane]      = (__bf16)cx;
  ar[128 + lane + 32] = (__bf16)cy;
  ar[128 + lane + 64] = (__bf16)cz;
  ar[128 + lane + 96] = (__bf16)cw;
}

// ---------------------------------------------------------------------------
// K5: out = Abf @ fc_wT' + fc_b   (M=4096, N=32000, K=256)  -- roofline stage:
// 524 MB fp32 out @ 23.3 TB/s ~ 23 us.  A(32x256) + B(64x256) bf16 tiles staged
// in LDS via TDM tensor_load_to_lds, s_wait_tensorcnt, then 8 waves x one
// 16x16 WMMA tile each; C repacked through LDS for coalesced float4 stores.
// ---------------------------------------------------------------------------
__global__ void fc_gemm_kernel(const __bf16* __restrict__ Abf,
                               const __bf16* __restrict__ fcwT,
                               const float* __restrict__ fc_b,
                               float* __restrict__ out) {
  __shared__ __align__(16) __bf16 a_lds[32 * 256];   // 16 KB (reused as f32 out tile)
  __shared__ __align__(16) __bf16 b_lds[64 * 256];   // 32 KB
  const int lane = threadIdx.x & 31;
  const int wave = threadIdx.x >> 5;
  const int half = lane >> 4;
  const int nbks = V_ / 64;                  // 500
  const int nb = blockIdx.x % nbks;
  const int mt = blockIdx.x / nbks;          // 128 M-blocks of 32
  const int m0 = mt * 32;
  const int n0 = nb * 64;

#if __has_builtin(__builtin_amdgcn_tensor_load_to_lds)
  if (wave == 0) {
    // A: rows m0..m0+31 of Abf (4096 x 256 bf16)
    tdm_load_2d_bf16(lds_offset_of(a_lds), Abf + (long long)m0 * 256,
                     256u, (unsigned)(B_ * T_), 256u, 32u, 256u);
    // B: rows n0..n0+63 of fcwT (32000 x 256 bf16)
    tdm_load_2d_bf16(lds_offset_of(b_lds), fcwT + (long long)n0 * 256,
                     256u, (unsigned)V_, 256u, 64u, 256u);
    __builtin_amdgcn_s_wait_tensorcnt(0);
  }
#else
  {
    const uint4* ga = (const uint4*)(Abf + (long long)m0 * 256);
    uint4* la = (uint4*)a_lds;
    for (int j = threadIdx.x; j < 32 * 256 / 8; j += blockDim.x) la[j] = ga[j];
    const uint4* gb = (const uint4*)(fcwT + (long long)n0 * 256);
    uint4* lb = (uint4*)b_lds;
    for (int j = threadIdx.x; j < 64 * 256 / 8; j += blockDim.x) lb[j] = gb[j];
  }
#endif
  __syncthreads();

  const int mtile = wave >> 2;               // 0..1
  const int ntile = wave & 3;                // 0..3
  const __bf16* arow = a_lds + (mtile * 16 + (lane & 15)) * 256;
  const __bf16* brow = b_lds + (ntile * 16 + (lane & 15)) * 256;

  v8f c = {};
#pragma unroll
  for (int kb = 0; kb < 256; kb += 32) {
    v16bf a = load_afrag_bf(arow, kb, half);
    v16bf b = load_bfrag(brow + kb + half * 16);
    c = wmma_bf16(a, b, c);
  }
  __syncthreads();                           // done reading a_lds; reuse as f32

  float* o_lds = (float*)a_lds;              // 32x64 f32 = 8 KB
#pragma unroll
  for (int r = 0; r < 8; ++r) {
    int mi = r + 8 * half;
    o_lds[(mtile * 16 + mi) * 64 + ntile * 16 + (lane & 15)] = c[r];
  }
  __syncthreads();

  // coalesced float4 epilogue with bias
  const float4* bias4 = (const float4*)(fc_b + n0);
  for (int j = threadIdx.x; j < 32 * 64 / 4; j += blockDim.x) {
    int row = j >> 4;                        // 16 float4 per row
    int c4  = j & 15;
    float4 v = ((const float4*)o_lds)[j];
    float4 bv = bias4[c4];
    v.x += bv.x; v.y += bv.y; v.z += bv.z; v.w += bv.w;
    *(float4*)(out + (size_t)(m0 + row) * V_ + n0 + c4 * 4) = v;
  }
}

// ---------------------------------------------------------------------------
extern "C" void kernel_launch(void* const* d_in, const int* in_sizes, int n_in,
                              void* d_out, int out_size, void* d_ws, size_t ws_size,
                              hipStream_t stream) {
  const int*   x    = (const int*)  d_in[0];
  const float* emb  = (const float*)d_in[1];
  const float* Wih  = (const float*)d_in[2];
  const float* Whh  = (const float*)d_in[3];
  const float* bias = (const float*)d_in[4];
  const float* aw   = (const float*)d_in[5];
  const float* ab   = (const float*)d_in[6];
  const float* vw   = (const float*)d_in[7];
  const float* vb   = (const float*)d_in[8];
  const float* fcw  = (const float*)d_in[9];
  const float* fcb  = (const float*)d_in[10];
  float* out = (float*)d_out;

  // workspace partition (~28.2 MB)
  char* p = (char*)d_ws;
  float*  xg     = (float*)p;              p += (size_t)B_ * T_ * G3 * 4;   // 6.29 MB
  float*  hs     = (float*)p;              p += (size_t)B_ * T_ * H_ * 4;   // 2.10 MB
  __bf16* hsb    = (__bf16*)p;             p += (size_t)B_ * T_ * H_ * 2;   // 1.05 MB
  float*  scores = (float*)p;              p += (size_t)B_ * T_ * 4;        // 16 KB
  __bf16* Abf    = (__bf16*)p;             p += (size_t)B_ * T_ * 256 * 2;  // 2.10 MB
  __bf16* WihT   = (__bf16*)p;             p += (size_t)G3 * E_ * 2;        // 96 KB
  __bf16* WhhT   = (__bf16*)p;             p += (size_t)G3 * H_ * 2;        // 96 KB
  __bf16* awT    = (__bf16*)p;             p += (size_t)H_ * H_ * 2;        // 32 KB
  __bf16* fcwT   = (__bf16*)p;             p += (size_t)V_ * 256 * 2;       // 16.4 MB

  convert_small_kernel<<<(G3 * E_) / 256, 256, 0, stream>>>(Wih, Whh, aw,
                                                            WihT, WhhT, awT);
  convert_fcw_kernel<<<(V_ * 256) / 256, 256, 0, stream>>>(fcw, fcwT);
  embed_gates_kernel<<<(B_ * T_ / 16) * (G3 / 16) / 4, 128, 0, stream>>>(
      x, emb, WihT, bias, xg);
  gru_scan_kernel<<<B_ / 16, 256, 0, stream>>>(xg, WhhT, hs, hsb);
  attn_score_kernel<<<B_ * T_ / 16, 256, 0, stream>>>(hsb, awT, ab, vw, vb, scores);
  attn_context_kernel<<<B_ * T_ / 8, 256, 0, stream>>>(scores, hs, hsb, Abf);
  fc_gemm_kernel<<<(B_ * T_ / 32) * (V_ / 64), 256, 0, stream>>>(
      Abf, fcwT, fcb, out);
}
